// MambaBlock_13357348290688
// MI455X (gfx1250) — matrix operand order repaired and checked by reading, not compiled, not run on hardware
//
#include <hip/hip_runtime.h>

typedef __attribute__((ext_vector_type(16))) _Float16 v16h;
typedef __attribute__((ext_vector_type(8)))  float    v8f;

#define NBATCH 8
#define LSEQ   2048
#define TOK    (NBATCH*LSEQ)   // 16384 tokens
#define DMODEL 256
#define DIN    512
#define NSTATE 16
#define DCONVK 4
#define DTRANK 16

// ---------------------------------------------------------------------------
// A-fragment (16x32 f16, MxK) lane/element mapping per CDNA5 ISA 7.12.2:
//   lanes 0-15 : M=lane,    v0..3 hold K=0..7,  v4..7 hold K=16..23
//   lanes 16-31: M=lane-16, v0..3 hold K=8..15, v4..7 hold K=24..31
__device__ __forceinline__ int a_frag_pos(int m, int kk) {
    int hb  = (kk >> 3) & 1;
    int rem = kk - 8 * hb;              // in {0..7} U {16..23}
    int e   = (rem >= 16) ? (rem - 8) : rem;
    return (m + 16 * hb) * 16 + e;
}

// ---------------------------------------------------------------------------
// Pack fp32 weights W[Kreal x N] (row major) into f16 B-fragments:
// dst[((kt*Ntiles + ntile)*32 + lane)*16 + e] = W[kt*32 + 8*(lane>>4) + e + (e>=8 ? 8 : 0)][ntile*16 + (lane&15)]
__global__ void pack_b_kernel(const float* __restrict__ W, _Float16* __restrict__ dst,
                              int Kreal, int N, int Ntiles) {
    int idx  = blockIdx.x * blockDim.x + threadIdx.x;   // one thread per fragment-lane
    int lane = idx & 31;
    int rest = idx >> 5;
    int ntile = rest % Ntiles;
    int kt    = rest / Ntiles;
    int n  = ntile * 16 + (lane & 15);
    int kb = kt * 32 + 8 * (lane >> 4);
    v16h frag;
#pragma unroll
    for (int e = 0; e < 16; ++e) {
        int k = kb + e + ((e >= 8) ? 8 : 0);   // table order: K=8h+e, 16+8h+(e-8)
        float v = (k < Kreal) ? W[(size_t)k * N + n] : 0.f;
        frag[e] = (_Float16)v;
    }
    *(v16h*)(dst + (size_t)idx * 16) = frag;
}

// ---------------------------------------------------------------------------
// LayerNorm over DM=256, one wave per token, writes h directly as f16 A-fragments.
__global__ void ln_pack_a_kernel(const float* __restrict__ x, const float* __restrict__ gamma,
                                 const float* __restrict__ beta, _Float16* __restrict__ h_swz) {
    int wid  = threadIdx.x >> 5;
    int lane = threadIdx.x & 31;
    int t    = blockIdx.x * 8 + wid;            // blockDim = 256 -> 8 waves
    const float* xr = x + (size_t)t * DMODEL;
    float vals[8];
    float s = 0.f, s2 = 0.f;
    int c0 = lane * 8;
#pragma unroll
    for (int i = 0; i < 8; ++i) { float v = xr[c0 + i]; vals[i] = v; s += v; s2 += v * v; }
#pragma unroll
    for (int off = 16; off >= 1; off >>= 1) {
        s  += __shfl_xor(s,  off, 32);
        s2 += __shfl_xor(s2, off, 32);
    }
    float mu  = s * (1.f / DMODEL);
    float var = s2 * (1.f / DMODEL) - mu * mu;
    float inv = rsqrtf(var + 1e-5f);
    int mtile = t >> 4, m = t & 15;
    _Float16* base = h_swz + (size_t)mtile * (DMODEL / 32) * 512;
#pragma unroll
    for (int i = 0; i < 8; ++i) {
        int c = c0 + i;
        float hv = (vals[i] - mu) * inv * gamma[c] + beta[c];
        base[(c >> 5) * 512 + a_frag_pos(m, c & 31)] = (_Float16)hv;
    }
}

// ---------------------------------------------------------------------------
// Register-blocked WMMA macro-tile: MB x NB grid of 16x16 tiles per wave.
// Per K-step: MB A-fragment loads + NB B-fragment loads feed MB*NB WMMAs.
template <int MB, int NB>
__device__ __forceinline__ void wmma_block(const _Float16* __restrict__ A,
                                           const _Float16* __restrict__ B,
                                           int mtile0, int ntile0, int Kt, int Ntiles,
                                           int lane, v8f acc[MB][NB]) {
#pragma unroll
    for (int i = 0; i < MB; ++i)
#pragma unroll
        for (int j = 0; j < NB; ++j)
            acc[i][j] = (v8f){};
    for (int kt = 0; kt < Kt; ++kt) {
        v16h a[MB];
#pragma unroll
        for (int i = 0; i < MB; ++i)
            a[i] = *(const v16h*)(A + (((size_t)(mtile0 + i) * Kt + kt) * 32 + lane) * 16);
#pragma unroll
        for (int j = 0; j < NB; ++j) {
            v16h b = *(const v16h*)(B + (((size_t)kt * Ntiles + (ntile0 + j)) * 32 + lane) * 16);
#pragma unroll
            for (int i = 0; i < MB; ++i)
                acc[i][j] = __builtin_amdgcn_wmma_f32_16x16x32_f16(false, a[i], false, b,
                                                                   (short)0, acc[i][j],
                                                                   false, false);
        }
    }
}

// GEMM1: xz[TOK,1024] = h @ W_in    (2x4 macro-tile)
__global__ void gemm_xz_kernel(const _Float16* __restrict__ A, const _Float16* __restrict__ B,
                               float* __restrict__ C) {
    int wid  = (blockIdx.x * blockDim.x + threadIdx.x) >> 5;
    int lane = threadIdx.x & 31;
    const int Ntiles = 1024 / 16, Kt = DMODEL / 32, MB = 2, NB = 4;
    const int nblk = Ntiles / NB;
    int mtile0 = (wid / nblk) * MB;
    int ntile0 = (wid % nblk) * NB;
    v8f acc[MB][NB];
    wmma_block<MB, NB>(A, B, mtile0, ntile0, Kt, Ntiles, lane, acc);
    int coff = lane & 15;
    int roff = 8 * (lane >> 4);
#pragma unroll
    for (int i = 0; i < MB; ++i)
#pragma unroll
        for (int j = 0; j < NB; ++j) {
            int col  = (ntile0 + j) * 16 + coff;
            int mrow = (mtile0 + i) * 16 + roff;
#pragma unroll
            for (int r = 0; r < 8; ++r)
                C[(size_t)(mrow + r) * 1024 + col] = acc[i][j][r];
        }
}

// Depthwise causal conv (DCONV=4) + bias + SiLU; writes fp32 xc and f16 A-fragments.
__global__ void conv_silu_kernel(const float* __restrict__ xz, const float* __restrict__ cw,
                                 const float* __restrict__ cb, float* __restrict__ xc,
                                 _Float16* __restrict__ xc_swz) {
    int idx = blockIdx.x * blockDim.x + threadIdx.x;     // TOK*DIN
    int d = idx & (DIN - 1);
    int t = idx >> 9;
    int l = t & (LSEQ - 1);
    int b = t >> 11;
    float acc = cb[d];
#pragma unroll
    for (int k = 0; k < DCONVK; ++k) {
        int j = l - (DCONVK - 1) + k;
        if (j >= 0)
            acc += xz[(size_t)(b * LSEQ + j) * 1024 + d] * cw[d * DCONVK + k];
    }
    float sv = acc / (1.f + __expf(-acc));
    xc[idx] = sv;
    int mtile = t >> 4, m = t & 15;
    xc_swz[((size_t)mtile * (DIN / 32) + (d >> 5)) * 512 + a_frag_pos(m, d & 31)] = (_Float16)sv;
}

// GEMM2: dbc[TOK,48] = xc @ W_x   (4x1 macro-tile since Ntiles=3)
__global__ void gemm_dbc_kernel(const _Float16* __restrict__ A, const _Float16* __restrict__ B,
                                float* __restrict__ C) {
    int wid  = (blockIdx.x * blockDim.x + threadIdx.x) >> 5;
    int lane = threadIdx.x & 31;
    const int Ntiles = 3, Kt = DIN / 32, MB = 4;
    int mtile0 = (wid / Ntiles) * MB;
    int ntile  = wid % Ntiles;
    v8f acc[MB][1];
    wmma_block<MB, 1>(A, B, mtile0, ntile, Kt, Ntiles, lane, acc);
    int col  = ntile * 16 + (lane & 15);
    int roff = 8 * (lane >> 4);
#pragma unroll
    for (int i = 0; i < MB; ++i) {
        int mrow = (mtile0 + i) * 16 + roff;
#pragma unroll
        for (int r = 0; r < 8; ++r)
            C[(size_t)(mrow + r) * 48 + col] = acc[i][0][r];
    }
}

// Pack dt_r = dbc[:, :16] into zero-padded (K=32) f16 A-fragments.
__global__ void pack_dt_kernel(const float* __restrict__ dbc, _Float16* __restrict__ dtr_swz) {
    int idx  = blockIdx.x * blockDim.x + threadIdx.x;    // (TOK/16)*32
    int lane = idx & 31;
    int mtile = idx >> 5;
    int m  = lane & 15;
    int hb = lane >> 4;
    v16h frag;
#pragma unroll
    for (int e = 0; e < 16; ++e) {
        int k = 8 * hb + e + ((e >= 8) ? 8 : 0);
        frag[e] = (k < DTRANK) ? (_Float16)dbc[(size_t)(mtile * 16 + m) * 48 + k]
                               : (_Float16)0.f;
    }
    *(v16h*)(dtr_swz + (size_t)idx * 16) = frag;
}

// GEMM3: dt[TOK,512] = softplus(dt_r @ W_dt + b_dt)   (2x4 macro-tile, Kt=1)
__global__ void gemm_dt_kernel(const _Float16* __restrict__ A, const _Float16* __restrict__ B,
                               const float* __restrict__ b_dt, float* __restrict__ dt) {
    int wid  = (blockIdx.x * blockDim.x + threadIdx.x) >> 5;
    int lane = threadIdx.x & 31;
    const int Ntiles = DIN / 16, Kt = 1, MB = 2, NB = 4;
    const int nblk = Ntiles / NB;
    int mtile0 = (wid / nblk) * MB;
    int ntile0 = (wid % nblk) * NB;
    v8f acc[MB][NB];
    wmma_block<MB, NB>(A, B, mtile0, ntile0, Kt, Ntiles, lane, acc);
    int coff = lane & 15;
    int roff = 8 * (lane >> 4);
#pragma unroll
    for (int i = 0; i < MB; ++i)
#pragma unroll
        for (int j = 0; j < NB; ++j) {
            int col  = (ntile0 + j) * 16 + coff;
            float bv = b_dt[col];
            int mrow = (mtile0 + i) * 16 + roff;
#pragma unroll
            for (int r = 0; r < 8; ++r) {
                float v = acc[i][j][r] + bv;
                v = (v > 20.f) ? v : log1pf(__expf(v));
                dt[(size_t)(mrow + r) * DIN + col] = v;
            }
        }
}

// Selective scan: one lane per (b,d); 16-state recurrence over L, fused
// D-residual + silu(z) gate; emits f16 A-fragments for the output GEMM.
__global__ void scan_kernel(const float* __restrict__ dt, const float* __restrict__ xc,
                            const float* __restrict__ dbc, const float* __restrict__ xz,
                            const float* __restrict__ A_log, const float* __restrict__ Dp,
                            _Float16* __restrict__ yact_swz) {
    int idx = blockIdx.x * blockDim.x + threadIdx.x;     // NBATCH*DIN = 4096
    int d = idx & (DIN - 1);
    int b = idx >> 9;
    float An[NSTATE], hst[NSTATE];
#pragma unroll
    for (int n = 0; n < NSTATE; ++n) {
        An[n]  = -__expf(A_log[d * NSTATE + n]);
        hst[n] = 0.f;
    }
    float Dd = Dp[d];
    for (int l = 0; l < LSEQ; ++l) {
        int t = b * LSEQ + l;
        float dtv = dt[(size_t)t * DIN + d];
        float xcv = xc[(size_t)t * DIN + d];
        float zv  = xz[(size_t)t * 1024 + DIN + d];
        const float4* Bp = (const float4*)(dbc + (size_t)t * 48 + DTRANK);
        const float4* Cp = (const float4*)(dbc + (size_t)t * 48 + DTRANK + NSTATE);
        float y = 0.f;
#pragma unroll
        for (int q = 0; q < 4; ++q) {
            float4 Bq = Bp[q];
            float4 Cq = Cp[q];
            float bb[4] = {Bq.x, Bq.y, Bq.z, Bq.w};
            float cc[4] = {Cq.x, Cq.y, Cq.z, Cq.w};
#pragma unroll
            for (int j = 0; j < 4; ++j) {
                int n = q * 4 + j;
                float dA = __expf(dtv * An[n]);
                hst[n] = dA * hst[n] + dtv * bb[j] * xcv;
                y += hst[n] * cc[j];
            }
        }
        float yv = (y + xcv * Dd) * (zv / (1.f + __expf(-zv)));
        int mtile = t >> 4, m = t & 15;
        yact_swz[((size_t)mtile * (DIN / 32) + (d >> 5)) * 512 + a_frag_pos(m, d & 31)]
            = (_Float16)yv;
    }
}

// GEMM4: out[TOK,256] = x + yact @ W_out   (2x4 macro-tile)
__global__ void gemm_out_kernel(const _Float16* __restrict__ A, const _Float16* __restrict__ B,
                                const float* __restrict__ x, float* __restrict__ out) {
    int wid  = (blockIdx.x * blockDim.x + threadIdx.x) >> 5;
    int lane = threadIdx.x & 31;
    const int Ntiles = DMODEL / 16, Kt = DIN / 32, MB = 2, NB = 4;
    const int nblk = Ntiles / NB;
    int mtile0 = (wid / nblk) * MB;
    int ntile0 = (wid % nblk) * NB;
    v8f acc[MB][NB];
    wmma_block<MB, NB>(A, B, mtile0, ntile0, Kt, Ntiles, lane, acc);
    int coff = lane & 15;
    int roff = 8 * (lane >> 4);
#pragma unroll
    for (int i = 0; i < MB; ++i)
#pragma unroll
        for (int j = 0; j < NB; ++j) {
            int col  = (ntile0 + j) * 16 + coff;
            int mrow = (mtile0 + i) * 16 + roff;
#pragma unroll
            for (int r = 0; r < 8; ++r) {
                size_t o = (size_t)(mrow + r) * DMODEL + col;
                out[o] = x[o] + acc[i][j][r];
            }
        }
}

// ---------------------------------------------------------------------------
extern "C" void kernel_launch(void* const* d_in, const int* in_sizes, int n_in,
                              void* d_out, int out_size, void* d_ws, size_t ws_size,
                              hipStream_t stream) {
    const float* x      = (const float*)d_in[0];
    const float* gamma  = (const float*)d_in[1];
    const float* beta   = (const float*)d_in[2];
    const float* W_in   = (const float*)d_in[3];
    const float* conv_w = (const float*)d_in[4];
    const float* conv_b = (const float*)d_in[5];
    const float* W_x    = (const float*)d_in[6];
    const float* W_dt   = (const float*)d_in[7];
    const float* b_dt   = (const float*)d_in[8];
    const float* A_log  = (const float*)d_in[9];
    const float* Dp     = (const float*)d_in[10];
    const float* W_out  = (const float*)d_in[11];
    float* out = (float*)d_out;

    char* p = (char*)d_ws;
    auto carve = [&](size_t bytes) -> void* {
        void* r = (void*)p;
        p += (bytes + 255) & ~(size_t)255;
        return r;
    };
    _Float16* h_swz    = (_Float16*)carve((size_t)TOK * DMODEL * 2);
    _Float16* win_swz  = (_Float16*)carve((size_t)DMODEL * 1024 * 2);      // Kt=8  Nt=64
    _Float16* wx_swz   = (_Float16*)carve((size_t)16 * 3 * 512 * 2);       // Kt=16 Nt=3
    _Float16* wdt_swz  = (_Float16*)carve((size_t)1 * 32 * 512 * 2);       // Kt=1  Nt=32 (K padded 16->32)
    _Float16* wout_swz = (_Float16*)carve((size_t)16 * 16 * 512 * 2);      // Kt=16 Nt=16
    float*    xz       = (float*)carve((size_t)TOK * 1024 * 4);
    float*    xc       = (float*)carve((size_t)TOK * DIN * 4);
    _Float16* xc_swz   = (_Float16*)carve((size_t)TOK * DIN * 2);
    float*    dbc      = (float*)carve((size_t)TOK * 48 * 4);
    _Float16* dtr_swz  = (_Float16*)carve((size_t)(TOK / 16) * 32 * 16 * 2);
    float*    dtb      = (float*)carve((size_t)TOK * DIN * 4);
    _Float16* yact_swz = (_Float16*)carve((size_t)TOK * DIN * 2);
    (void)ws_size; (void)in_sizes; (void)n_in; (void)out_size;

    const int BT = 256;
    // Weight packing (f32 -> f16 B-fragments)
    pack_b_kernel<<<(8 * 64 * 32) / BT, BT, 0, stream>>>(W_in,  win_swz,  DMODEL, 1024, 64);
    pack_b_kernel<<<(16 * 3 * 32) / BT, BT, 0, stream>>>(W_x,   wx_swz,   DIN,    48,   3);
    pack_b_kernel<<<(1 * 32 * 32) / BT, BT, 0, stream>>>(W_dt,  wdt_swz,  DTRANK, DIN,  32);
    pack_b_kernel<<<(16 * 16 * 32) / BT, BT, 0, stream>>>(W_out, wout_swz, DIN,    DMODEL, 16);

    // LayerNorm -> f16 A-fragments
    ln_pack_a_kernel<<<TOK / 8, BT, 0, stream>>>(x, gamma, beta, h_swz);

    // xz = h @ W_in   (16384x1024x256): (1024/2)*(64/4) = 8192 waves
    gemm_xz_kernel<<<(8192 * 32) / BT, BT, 0, stream>>>(h_swz, win_swz, xz);

    // depthwise conv + SiLU
    conv_silu_kernel<<<(TOK * DIN) / BT, BT, 0, stream>>>(xz, conv_w, conv_b, xc, xc_swz);

    // dbc = xc @ W_x  (16384x48x512): (1024/4)*3 = 768 waves
    gemm_dbc_kernel<<<(768 * 32) / BT, BT, 0, stream>>>(xc_swz, wx_swz, dbc);

    // dt_r -> padded f16 A-fragments
    pack_dt_kernel<<<((TOK / 16) * 32) / BT, BT, 0, stream>>>(dbc, dtr_swz);

    // dt = softplus(dt_r @ W_dt + b_dt)  (16384x512x32): (1024/2)*(32/4) = 4096 waves
    gemm_dt_kernel<<<(4096 * 32) / BT, BT, 0, stream>>>(dtr_swz, wdt_swz, b_dt, dtb);

    // selective scan + gate
    scan_kernel<<<(NBATCH * DIN) / BT, BT, 0, stream>>>(dtb, xc, dbc, xz, A_log, Dp, yact_swz);

    // out = x + yact @ W_out  (16384x256x512): (1024/2)*(16/4) = 2048 waves
    gemm_out_kernel<<<(2048 * 32) / BT, BT, 0, stream>>>(yact_swz, wout_swz, x, out);
}
